// GriffinRecurrence_37177236914305
// MI455X (gfx1250) — matrix operand order repaired and checked by reading, not compiled, not run on hardware
//
#include <hip/hip_runtime.h>
#include <hip/hip_bf16.h>

// ---------------------------------------------------------------------------
// Types for WMMA (gfx1250, wave32)
// ---------------------------------------------------------------------------
typedef __attribute__((ext_vector_type(8)))  float          v8f;
typedef __attribute__((ext_vector_type(16))) __bf16         v16bf;
typedef __attribute__((ext_vector_type(8)))  unsigned short v8u;
typedef __attribute__((ext_vector_type(16))) unsigned short v16u;

#define B_SZ    4
#define S_SZ    4096
#define DM      1024
#define DR      384
#define CHK     64
#define NCHK    (S_SZ / CHK)          // 64
#define NTOK    (B_SZ * S_SZ)         // 16384
#define NCHUNKS (NTOK / CHK)          // 256 workgroups for the GEMM
#define WELEM   (DR * DM)             // elements per weight matrix

// LDS x-panel: 64 rows x 256 K (bf16), padded row stride to dodge bank conflicts
#define KPAN    256
#define XSTR    264                    // 264*2B = 528B row stride; 528/4 dwords = 132 ≡ 4 (mod 64)

static __device__ __forceinline__ unsigned short f32_to_bf16(float f) {
    unsigned int u = __float_as_uint(f);
    unsigned int r = u + 0x7FFFu + ((u >> 16) & 1u);   // round-to-nearest-even
    return (unsigned short)(r >> 16);
}

static __device__ __forceinline__ float sigmoidf(float x) {
    return 1.0f / (1.0f + __expf(-x));
}

// ---------------------------------------------------------------------------
// Kernel 0: convert the three fp32 weight matrices to bf16 (concatenated)
// ---------------------------------------------------------------------------
__global__ void cvt_w(const float* __restrict__ wa, const float* __restrict__ wi,
                      const float* __restrict__ wv, unsigned short* __restrict__ wbf) {
    int i = blockIdx.x * blockDim.x + threadIdx.x;
    if (i < WELEM) {
        wbf[i]             = f32_to_bf16(wa[i]);
        wbf[i + WELEM]     = f32_to_bf16(wi[i]);
        wbf[i + 2 * WELEM] = f32_to_bf16(wv[i]);
    }
}

// ---------------------------------------------------------------------------
// Kernel 1: three fused GEMMs via WMMA + gating epilogue.
// grid = 256 (one 64-token chunk each), block = 512 (16 waves).
// wave w: row-block rb = w>>2 (16 tokens), col-quarter cq = w&3 (96 channels).
// Emits a[m,n] and u[m,n] (fp32) into workspace.
// ---------------------------------------------------------------------------
__global__ void __launch_bounds__(512, 1)
gemm_gates(const float* __restrict__ x, const unsigned short* __restrict__ wbf,
           const float* __restrict__ bias, float* __restrict__ out_a,
           float* __restrict__ out_u) {
    __shared__ __align__(16) unsigned short xs[64 * XSTR];

    const int tid   = threadIdx.x;
    const int chunk = blockIdx.x;              // 0..255  (token block of 64)
    const int wave  = tid >> 5;
    const int lane  = tid & 31;
    const int rb    = wave >> 2;               // 0..3
    const int cq    = wave & 3;                // 0..3
    const int lhalf = lane >> 4;               // 0/1
    const int llo   = lane & 15;

    v8f acc[3][6] = {};                        // [matrix][n-tile] fp32 accumulators

    for (int p = 0; p < DM / KPAN; ++p) {      // 4 K-panels of 256
        __syncthreads();
        // ---- stage 64x256 fp32 panel of x into LDS as bf16 (coalesced) ----
        #pragma unroll
        for (int i = 0; i < 32; ++i) {
            int e   = tid + i * 512;           // 0..16383
            int row = e >> 8;
            int k   = e & 255;
            float f = x[(chunk * CHK + row) * DM + p * KPAN + k];
            xs[row * XSTR + k] = f32_to_bf16(f);
        }
        __syncthreads();

        for (int kk = 0; kk < KPAN / 32; ++kk) {   // 8 WMMA K-steps per panel
            const int kb = kk * 32;
            // A fragment (16x32 bf16): lanes 0-15 hold K=0..15,
            // lanes 16-31 hold K=16..31 for row M = lane&15.
            const int arow = rb * 16 + llo;
            const int ksel = lhalf * 16;
            v8u lo = *(const v8u*)&xs[arow * XSTR + kb + ksel];
            v8u hi = *(const v8u*)&xs[arow * XSTR + kb + 8 + ksel];
            v16u af;
            #pragma unroll
            for (int i = 0; i < 8; ++i) { af[i] = lo[i]; af[8 + i] = hi[i]; }
            v16bf A = __builtin_bit_cast(v16bf, af);

            const int kg = p * KPAN + kb + lhalf * 16;  // global K base for B frag
            #pragma unroll
            for (int m = 0; m < 3; ++m) {
                const unsigned short* W = wbf + m * WELEM;
                #pragma unroll
                for (int t = 0; t < 6; ++t) {
                    // B fragment (32x16 bf16): lane group 0 holds K=0..15,
                    // group 1 holds K=16..31, for column N = lane&15.
                    const int n = cq * 96 + t * 16 + llo;
                    v16u bw = *(const v16u*)&W[n * DM + kg];
                    v16bf Bf = __builtin_bit_cast(v16bf, bw);
                    acc[m][t] = __builtin_amdgcn_wmma_f32_16x16x32_bf16(
                        false, A, false, Bf, (short)0, acc[m][t], false, false);
                }
            }
        }
    }

    // ---- fused gating epilogue -------------------------------------------
    // D layout: element r of v8f is row (r + 8*(lane>>4)), column lane&15.
    #pragma unroll
    for (int t = 0; t < 6; ++t) {
        const int n  = cq * 96 + t * 16 + llo;
        const float bb = bias[n];
        #pragma unroll
        for (int r = 0; r < 8; ++r) {
            const int m  = chunk * CHK + rb * 16 + r + lhalf * 8;
            float a  = sigmoidf(acc[0][t][r] + bb);
            float ig = sigmoidf(acc[1][t][r]);
            float vv = acc[2][t][r];
            float u  = sqrtf(fmaxf(1.0f - a * a, 1e-8f)) * ig * vv;
            out_a[m * DR + n] = a;
            out_u[m * DR + n] = u;
        }
    }
}

// ---------------------------------------------------------------------------
// Kernel 2: intra-chunk scan. One thread per (b, chunk, d) = 98304 threads.
// Writes provisional states to out, chunk decay-product to cp, last state to cl.
// ---------------------------------------------------------------------------
__global__ void scan_intra(const float* __restrict__ wa, const float* __restrict__ wu,
                           float* __restrict__ out, float* __restrict__ cp,
                           float* __restrict__ cl) {
    int gid = blockIdx.x * blockDim.x + threadIdx.x;    // 0..98303
    int d  = gid % DR;
    int bc = gid / DR;                                  // b*64 + c
    int c  = bc % NCHK;
    int b  = bc / NCHK;
    int base = ((b * S_SZ) + c * CHK) * DR + d;
    float h = 0.0f, pr = 1.0f;
    for (int t = 0; t < CHK; ++t) {
        float a = wa[base + t * DR];
        float u = wu[base + t * DR];
        h = fmaf(a, h, u);
        pr *= a;
        out[base + t * DR] = h;
    }
    cp[bc * DR + d] = pr;
    cl[bc * DR + d] = h;
}

// ---------------------------------------------------------------------------
// Kernel 3: inter-chunk carry scan. One thread per (b, d) = 1536 threads.
// carry[c] = state entering chunk c.
// ---------------------------------------------------------------------------
__global__ void scan_carry(const float* __restrict__ cp, const float* __restrict__ cl,
                           float* __restrict__ carry) {
    int gid = blockIdx.x * blockDim.x + threadIdx.x;    // 0..1535
    int d = gid % DR;
    int b = gid / DR;
    float s = 0.0f;
    for (int c = 0; c < NCHK; ++c) {
        int idx = (b * NCHK + c) * DR + d;
        carry[idx] = s;
        s = fmaf(cp[idx], s, cl[idx]);
    }
}

// ---------------------------------------------------------------------------
// Kernel 4: apply carry: out[t] += cumprod(a[0..t]) * carry(chunk).
// ---------------------------------------------------------------------------
__global__ void scan_apply(const float* __restrict__ wa, const float* __restrict__ carry,
                           float* __restrict__ out) {
    int gid = blockIdx.x * blockDim.x + threadIdx.x;
    int d  = gid % DR;
    int bc = gid / DR;
    int c  = bc % NCHK;
    int b  = bc / NCHK;
    float cr = carry[bc * DR + d];
    int base = ((b * S_SZ) + c * CHK) * DR + d;
    float p = 1.0f;
    for (int t = 0; t < CHK; ++t) {
        p *= wa[base + t * DR];
        out[base + t * DR] = fmaf(p, cr, out[base + t * DR]);
    }
}

// ---------------------------------------------------------------------------
// Host-side launch. Inputs: x, Wa, Wi, Wv, decay_bias (setup_inputs order).
// Workspace layout (bytes):
//   [0)                 bf16 weights  3*384*1024*2 = 2,359,296  (256-aligned end)
//   [2359296)           a             16384*384*4  = 25,165,824
//   [+]                 u             16384*384*4  = 25,165,824
//   [+]                 cp, cl, carry 3 * 98304*4  =  1,179,648
// total ~51.4 MB
// ---------------------------------------------------------------------------
extern "C" void kernel_launch(void* const* d_in, const int* in_sizes, int n_in,
                              void* d_out, int out_size, void* d_ws, size_t ws_size,
                              hipStream_t stream) {
    const float* x    = (const float*)d_in[0];
    const float* Wa   = (const float*)d_in[1];
    const float* Wi   = (const float*)d_in[2];
    const float* Wv   = (const float*)d_in[3];
    const float* bias = (const float*)d_in[4];
    float* out = (float*)d_out;

    char* wsb = (char*)d_ws;
    unsigned short* wbf = (unsigned short*)wsb;                 // 2,359,296 B
    float* ws_a  = (float*)(wsb + 2359296);
    float* ws_u  = ws_a + (size_t)NTOK * DR;
    float* cp    = ws_u + (size_t)NTOK * DR;
    float* cl    = cp + (size_t)B_SZ * NCHK * DR;
    float* carry = cl + (size_t)B_SZ * NCHK * DR;

    // 0) weights fp32 -> bf16
    cvt_w<<<(WELEM + 255) / 256, 256, 0, stream>>>(Wa, Wi, Wv, wbf);

    // 1) fused triple GEMM + gating (WMMA bf16)
    gemm_gates<<<NCHUNKS, 512, 0, stream>>>(x, wbf, bias, ws_a, ws_u);

    // 2) intra-chunk scan
    const int nscan = B_SZ * NCHK * DR;                          // 98304
    scan_intra<<<nscan / 256, 256, 0, stream>>>(ws_a, ws_u, out, cp, cl);

    // 3) inter-chunk carry
    scan_carry<<<(B_SZ * DR) / 256, 256, 0, stream>>>(cp, cl, carry);

    // 4) apply carry
    scan_apply<<<nscan / 256, 256, 0, stream>>>(ws_a, carry, out);
}
